// DCNv2Block_MG_80101140070677
// MI455X (gfx1250) — matrix operand order repaired and checked by reading, not compiled, not run on hardware
//
#include <hip/hip_runtime.h>
#include <hip/hip_bf16.h>

typedef __attribute__((ext_vector_type(16))) _Float16 v16h;
typedef __attribute__((ext_vector_type(2)))  _Float16 v2h;
typedef __attribute__((ext_vector_type(8)))  float    v8f;

#define HTOT   (8 * 128 * 4096)        // B*C*H*W elements
#define SPB    4096                    // H*W
#define NOFF   108                     // 3*K2*OG
#define PW     66                      // padded width/height
#define PAREA  (PW * PW)               // 4356
#define PADTOT (8 * PAREA * 128)       // 4,460,544 padded f16 elements

// ---------------- packing kernels ----------------

// x: NCHW f32 -> xh: zero-padded NHWC f16 [B][66][66][128]
__global__ __launch_bounds__(256) void pack_x_kernel(const float* __restrict__ x,
                                                     _Float16* __restrict__ xh) {
    int e = blockIdx.x * 256 + threadIdx.x;
    if (e >= PADTOT) return;
    int c  = e & 127;
    int t2 = e >> 7;               // b*4356 + yy*66 + xx
    int xx = t2 % PW;
    int t3 = t2 / PW;
    int yy = t3 % PW;
    int b  = t3 / PW;
    float v = 0.0f;
    if (yy >= 1 && yy <= 64 && xx >= 1 && xx <= 64)
        v = x[((size_t)(b * 128 + c)) * SPB + (yy - 1) * 64 + (xx - 1)];
    xh[e] = (_Float16)v;
}

// w_off (108,128,3,3) -> Bw[k][n] f16, k = r*128+c, n = oc (padded to 112)
__global__ __launch_bounds__(256) void pack_woff_kernel(const float* __restrict__ w,
                                                        _Float16* __restrict__ Bw) {
    int e = blockIdx.x * 256 + threadIdx.x;
    if (e >= 1152 * 112) return;
    int n = e % 112;
    int k = e / 112;
    int c = k & 127;
    int r = k >> 7;
    Bw[e] = (n < NOFF) ? (_Float16)w[((size_t)n * 128 + c) * 9 + r] : (_Float16)0.0f;
}

// w_dcn (128,128,3,3) -> Bd[k][n] f16, k = r*128+c, n = oc
__global__ __launch_bounds__(256) void pack_wdcn_kernel(const float* __restrict__ w,
                                                        _Float16* __restrict__ Bd) {
    int e = blockIdx.x * 256 + threadIdx.x;
    if (e >= 1152 * 128) return;
    int n = e & 127;
    int k = e >> 7;
    int c = k & 127;
    int r = k >> 7;
    Bd[e] = (_Float16)w[((size_t)n * 128 + c) * 9 + r];
}

// ---------------- WMMA fragment helper ----------------
// A (16x32 f16): lane l -> row l&15; halves 0..7 = K kb..kb+7,
//                halves 8..15 = K kb+16..kb+23, kb = (l>>4)*8.
__device__ __forceinline__ v16h load_a_frag(const _Float16* At, int ld, int lane) {
    int row = lane & 15;
    int kb  = (lane >> 4) << 3;
    const _Float16* p = At + row * ld + kb;
    v16h a;
#pragma unroll
    for (int j = 0; j < 8; ++j) a[j] = p[j];
#pragma unroll
    for (int j = 0; j < 8; ++j) a[j + 8] = p[16 + j];
    return a;
}

// async global -> LDS 128-bit copy (ASYNCcnt-tracked, no VGPR staging)
__device__ __forceinline__ void async_copy_b128(const _Float16* gsrc, _Float16* ldst) {
    unsigned           lds_addr = (unsigned)(size_t)ldst;      // flat[31:0] == LDS offset
    unsigned long long gaddr    = (unsigned long long)(size_t)gsrc;
    asm volatile("global_load_async_to_lds_b128 %0, %1, off"
                 :: "v"(lds_addr), "v"(gaddr) : "memory");
}

__device__ __forceinline__ void wait_async_zero() {
    asm volatile("s_wait_asynccnt 0x0" ::: "memory");
}

// ---------------- kernel 1: offset conv (implicit GEMM) ----------------
// grid = (M/128, 7); 8 waves, wave w owns M-subtile w, all share N-tile blockIdx.y
__global__ __launch_bounds__(256) void offset_conv_kernel(
    const _Float16* __restrict__ xh, const _Float16* __restrict__ Bw,
    const float* __restrict__ b_off, float* __restrict__ offm) {
    __shared__ _Float16 At[128 * 136];
    const int t    = threadIdx.x;
    const int lane = t & 31;
    const int wave = t >> 5;
    const int m0   = blockIdx.x * 128;
    const int n0   = blockIdx.y * 16;
    // fixed row assignment for the cooperative A-tile copy
    const int mloc = t >> 1, part = t & 1;
    const int m  = m0 + mloc;
    const int b  = m >> 12;
    const int sp = m & 4095;
    const int yy = sp >> 6, xx = sp & 63;
    const _Float16* rowbase =
        xh + ((size_t)b * PAREA + yy * PW + xx) * 128 + part * 64;
    _Float16* dst = &At[mloc * 136 + part * 64];

    v8f acc = {};
    for (int r = 0; r < 9; ++r) {
        const int ki = r / 3, kj = r % 3;
        const _Float16* src = rowbase + (ki * PW + kj) * 128;
        __syncthreads();
#pragma unroll
        for (int j = 0; j < 4; ++j) {
            async_copy_b128(src + j * 16,     dst + j * 16);
            async_copy_b128(src + j * 16 + 8, dst + j * 16 + 8);
        }
        wait_async_zero();
        __syncthreads();
        const int k0 = r * 128;
        if (r < 8)  // speculative prefetch of next tap's B rows
            __builtin_prefetch(Bw + (size_t)(k0 + 128 + lane) * 112 + n0, 0, 1);
#pragma unroll
        for (int cc = 0; cc < 4; ++cc) {
            v16h bf;
            const _Float16* bp = Bw + (size_t)(k0 + cc * 32 + ((lane >> 4) << 3)) * 112 + n0 + (lane & 15);
#pragma unroll
            for (int j = 0; j < 8; ++j) { bf[j] = bp[j * 112]; bf[j + 8] = bp[(16 + j) * 112]; }
            v16h af = load_a_frag(At + wave * 16 * 136 + cc * 32, 136, lane);
            acc = __builtin_amdgcn_wmma_f32_16x16x32_f16(false, af, false, bf,
                                                         (short)0, acc, false, false);
        }
    }
    // epilogue: bias + clip/sigmoid, scatter to offm[m][og][r][{dy,dx,mask}]
    const int col = lane & 15;
    const int n   = n0 + col;
    if (n < NOFF) {
        const float bias = b_off[n];
        const int half = (lane >> 4) << 3;
#pragma unroll
        for (int j = 0; j < 8; ++j) {
            int   mm = m0 + wave * 16 + half + j;
            float v  = acc[j] + bias;
            if (n < 72) {
                int og = n / 18, rr = (n % 18) >> 1, comp = n & 1;
                v = fminf(fmaxf(v, -64.0f), 64.0f);
                offm[((size_t)mm * 36 + og * 9 + rr) * 3 + comp] = v;
            } else {
                int tt = n - 72;
                int og = tt / 9, rr = tt % 9;
                v = 1.0f / (1.0f + __expf(-v));
                offm[((size_t)mm * 36 + og * 9 + rr) * 3 + 2] = v;
            }
        }
    }
}

// ---------------- kernel 2: deformable conv GEMM + GN partial stats ----------------
struct SDesc { int o00, o01, o10, o11; float w00, w01, w10, w11; };

// grid = M/16 ; 8 waves share one sampled A-tile, wave w owns N-tile w*16
__global__ __launch_bounds__(256) void deform_gemm_kernel(
    const _Float16* __restrict__ xh, const _Float16* __restrict__ Bd,
    const float* __restrict__ offm, float* __restrict__ yraw,
    float* __restrict__ stats) {
    __shared__ _Float16 At[16 * 136];
    __shared__ SDesc    sdesc[576];          // [mloc][og][r]
    __shared__ float    s_sum[32], s_ssq[32];
    const int t    = threadIdx.x;
    const int lane = t & 31;
    const int wave = t >> 5;
    const int m0   = blockIdx.x * 16;
    const int n0   = wave * 16;
    const int b    = m0 >> 12;
    if (t < 32) { s_sum[t] = 0.0f; s_ssq[t] = 0.0f; }

    // phase 0: bilinear sample descriptors, one per (mloc, og, r)
    for (int e = t; e < 576; e += 256) {
        int mloc = e / 36;
        int rem  = e - mloc * 36;
        int cc   = rem / 9;              // offset group
        int r    = rem - cc * 9;
        int m    = m0 + mloc;
        int sp   = m & 4095;
        int yy   = sp >> 6, xx = sp & 63;
        const float* om = &offm[((size_t)m * 36 + cc * 9 + r) * 3];
        float dy = om[0], dx = om[1], msk = om[2];
        int   ki = r / 3, kj = r % 3;
        float py = dy + (float)(ki + yy - 1);
        float px = dx + (float)(kj + xx - 1);
        float y0f = floorf(py), x0f = floorf(px);
        float wy1 = py - y0f, wx1 = px - x0f;
        float wy0 = 1.0f - wy1, wx0 = 1.0f - wx1;
        int y0 = (int)y0f, x0 = (int)x0f;
        // clamp into padded coords: all OOB corners land on the zero border
        int yp0 = min(max(y0 + 1, 0), 65), yp1 = min(max(y0 + 2, 0), 65);
        int xp0 = min(max(x0 + 1, 0), 65), xp1 = min(max(x0 + 2, 0), 65);
        SDesc d;
        d.o00 = (yp0 * PW + xp0) * 128;  d.o01 = (yp0 * PW + xp1) * 128;
        d.o10 = (yp1 * PW + xp0) * 128;  d.o11 = (yp1 * PW + xp1) * 128;
        d.w00 = wy0 * wx0 * msk;  d.w01 = wy0 * wx1 * msk;
        d.w10 = wy1 * wx0 * msk;  d.w11 = wy1 * wx1 * msk;
        sdesc[e] = d;
    }

    const _Float16* xb = xh + (size_t)b * PAREA * 128;
    const int amloc = t >> 4;            // 0..15 sampled row
    const int cp    = (t & 15) << 1;     // channel pair within group
    v8f acc = {};
    for (int r = 0; r < 9; ++r) {
        __syncthreads();
#pragma unroll
        for (int cc = 0; cc < 4; ++cc) {
            SDesc d = sdesc[(amloc * 4 + cc) * 9 + r];
            const _Float16* xc = xb + cc * 32 + cp;
            v2h h00 = *(const v2h*)(xc + d.o00);
            v2h h01 = *(const v2h*)(xc + d.o01);
            v2h h10 = *(const v2h*)(xc + d.o10);
            v2h h11 = *(const v2h*)(xc + d.o11);
            float lo = d.w00 * (float)h00[0] + d.w01 * (float)h01[0]
                     + d.w10 * (float)h10[0] + d.w11 * (float)h11[0];
            float hi = d.w00 * (float)h00[1] + d.w01 * (float)h01[1]
                     + d.w10 * (float)h10[1] + d.w11 * (float)h11[1];
            v2h o; o[0] = (_Float16)lo; o[1] = (_Float16)hi;
            *(v2h*)(&At[amloc * 136 + cc * 32 + cp]) = o;
        }
        __syncthreads();
        if (r < 8)
            __builtin_prefetch(Bd + (size_t)((r + 1) * 128 + lane) * 128 + n0, 0, 1);
#pragma unroll
        for (int cc = 0; cc < 4; ++cc) {
            v16h bf;
            const _Float16* bp = Bd + (size_t)(r * 128 + cc * 32 + ((lane >> 4) << 3)) * 128 + n0 + (lane & 15);
#pragma unroll
            for (int j = 0; j < 8; ++j) { bf[j] = bp[j * 128]; bf[j + 8] = bp[(16 + j) * 128]; }
            v16h af = load_a_frag(At + cc * 32, 136, lane);
            acc = __builtin_amdgcn_wmma_f32_16x16x32_f16(false, af, false, bf,
                                                         (short)0, acc, false, false);
        }
    }
    // write y (NCHW f32) + per-group partial sums
    const int col  = lane & 15;
    const int oc   = n0 + col;
    const int half = (lane >> 4) << 3;
    float ls = 0.0f, lss = 0.0f;
#pragma unroll
    for (int j = 0; j < 8; ++j) {
        int   sp = (m0 & 4095) + half + j;
        float v  = acc[j];
        yraw[(size_t)(b * 128 + oc) * SPB + sp] = v;
        ls += v;
        lss += v * v;
    }
    atomicAdd(&s_sum[oc >> 2], ls);
    atomicAdd(&s_ssq[oc >> 2], lss);
    __syncthreads();
    if (t < 32) {
        atomicAdd(&stats[(b * 32 + t) * 2 + 0], s_sum[t]);
        atomicAdd(&stats[(b * 32 + t) * 2 + 1], s_ssq[t]);
    }
}

// ---------------- kernel 3: GroupNorm finalize + affine + SiLU ----------------
__global__ __launch_bounds__(256) void gn_silu_kernel(
    const float* __restrict__ yraw, const float* __restrict__ stats,
    const float* __restrict__ gamma, const float* __restrict__ beta,
    float* __restrict__ out) {
    int e = blockIdx.x * 256 + threadIdx.x;
    if (e >= HTOT) return;
    int oc = (e >> 12) & 127;
    int b  = e >> 19;
    int g  = oc >> 2;
    float s    = stats[(b * 32 + g) * 2 + 0];
    float ss   = stats[(b * 32 + g) * 2 + 1];
    const float invN = 1.0f / 16384.0f;   // 4 ch * 4096 px per group
    float mu  = s * invN;
    float var = fmaxf(ss * invN - mu * mu, 0.0f);
    float rin = rsqrtf(var + 1e-5f);
    float yn  = (yraw[e] - mu) * rin * gamma[oc] + beta[oc];
    out[e] = yn / (1.0f + __expf(-yn));
}

// ---------------- host launch ----------------
extern "C" void kernel_launch(void* const* d_in, const int* in_sizes, int n_in,
                              void* d_out, int out_size, void* d_ws, size_t ws_size,
                              hipStream_t stream) {
    const float* x     = (const float*)d_in[0];
    const float* w_off = (const float*)d_in[1];
    const float* b_off = (const float*)d_in[2];
    const float* w_dcn = (const float*)d_in[3];
    const float* gamma = (const float*)d_in[4];
    const float* beta  = (const float*)d_in[5];
    float* out = (float*)d_out;

    char* ws = (char*)d_ws;
    _Float16* xh    = (_Float16*)(ws);                      //  8,921,088 B (padded NHWC)
    _Float16* Bw    = (_Float16*)(ws + 8921088);            //    258,048 B (pad 262,144)
    _Float16* Bd    = (_Float16*)(ws + 8921088 + 262144);   //    294,912 B
    float*    offm  = (float*)(ws + 9478144);               // 14,155,776 B
    float*    yraw  = (float*)(ws + 23633920);              // 16,777,216 B
    float*    stats = (float*)(ws + 40411136);              //      2,048 B

    hipMemsetAsync(stats, 0, 8 * 32 * 2 * sizeof(float), stream);

    pack_x_kernel<<<(PADTOT + 255) / 256, 256, 0, stream>>>(x, xh);
    pack_woff_kernel<<<(1152 * 112 + 255) / 256, 256, 0, stream>>>(w_off, Bw);
    pack_wdcn_kernel<<<(1152 * 128) / 256, 256, 0, stream>>>(w_dcn, Bd);

    dim3 g1(32768 / 128, 7);   // M-tiles x N-tiles(112/16)
    offset_conv_kernel<<<g1, 256, 0, stream>>>(xh, Bw, b_off, offm);

    deform_gemm_kernel<<<32768 / 16, 256, 0, stream>>>(xh, Bd, offm, yraw, stats);

    gn_silu_kernel<<<HTOT / 256, 256, 0, stream>>>(yraw, stats, gamma, beta, out);
}
